// MultiHeadSelfAttention_86002425135553
// MI455X (gfx1250) — matrix operand order repaired and checked by reading, not compiled, not run on hardware
//
#include <hip/hip_runtime.h>
#include <hip/hip_bf16.h>

typedef __attribute__((ext_vector_type(16))) _Float16 v16h;
typedef __attribute__((ext_vector_type(8)))  float    v8f;
typedef __attribute__((ext_vector_type(8)))  _Float16 h8;

#define WMMA_F32_F16(A, B, C) \
  __builtin_amdgcn_wmma_f32_16x16x32_f16(false, (A), false, (B), (short)0, (C), false, false)

// Build a 16-half fragment from two 16-byte chunks (A and B: p1=p0+16).
static __device__ __forceinline__ v16h frag_from(const _Float16* p0, const _Float16* p1) {
  h8 a = *(const h8*)p0;
  h8 b = *(const h8*)p1;
  v16h r;
#pragma unroll
  for (int i = 0; i < 8; ++i) { r[i] = a[i]; r[i + 8] = b[i]; }
  return r;
}

// ---------------------------------------------------------------------------
// Kernel 1: QKV projection + bias + RoPE.
//   grid = 32 row-blocks * 16 heads * 3 matrices = 1536 blocks of 256 (8 waves)
//   Each wave: 16 tokens x 64 cols (one head strip) -> 4 C-fragments.
//   Q,K -> [b,h,s,d] fp16 (RoPE applied). V -> [b,h,d,s] fp16 (transposed).
// ---------------------------------------------------------------------------
__global__ __launch_bounds__(256) void qkv_rope_kernel(
    const float* __restrict__ hidden,
    const float* __restrict__ wq, const float* __restrict__ wk, const float* __restrict__ wv,
    const float* __restrict__ bq, const float* __restrict__ bk, const float* __restrict__ bv,
    _Float16* __restrict__ qf, _Float16* __restrict__ kf, _Float16* __restrict__ vf) {
  __shared__ _Float16 wt[64 * 40];  // 64 cols x 32 k, padded stride 40 halves

  const int rb   = blockIdx.x & 31;
  const int h    = (blockIdx.x >> 5) & 15;
  const int mat  = blockIdx.x >> 9;  // 0=Q 1=K 2=V
  const int wave = threadIdx.x >> 5;
  const int lane = threadIdx.x & 31;
  const int rows = rb * 128 + wave * 16;

  const float* w    = (mat == 0) ? wq : (mat == 1) ? wk : wv;
  const float* bias = (mat == 0) ? bq : (mat == 1) ? bk : bv;

  v8f acc[4] = {};

  const int arow = rows + (lane & 15);
  const int aoff = (lane & 16) ? 8 : 0;

  for (int k0 = 0; k0 < 1024; k0 += 32) {
    __syncthreads();
    {  // stage W[k0..k0+32) x 64 cols, transposed+converted, into LDS
      const int nl = threadIdx.x & 63;
      const int kg = threadIdx.x >> 6;
#pragma unroll
      for (int i = 0; i < 8; ++i) {
        const int kl = kg * 8 + i;
        wt[nl * 40 + kl] = (_Float16)w[(size_t)(k0 + kl) * 1024 + h * 64 + nl];
      }
      if (k0 + 32 < 1024)
        __builtin_prefetch(&w[(size_t)(k0 + 32) * 1024 + h * 64 + nl], 0, 1);
    }
    __syncthreads();

    // A fragment: hidden (fp32 -> fp16), lane layout per ISA 16-bit A 16x32
    const float* ap = hidden + (size_t)arow * 1024 + k0 + aoff;
    v16h afrag;
#pragma unroll
    for (int i = 0; i < 8; ++i) { afrag[i] = (_Float16)ap[i]; afrag[i + 8] = (_Float16)ap[i + 16]; }

#pragma unroll
    for (int f = 0; f < 4; ++f) {
      const _Float16* bp = &wt[(f * 16 + (lane & 15)) * 40 + ((lane & 16) ? 8 : 0)];
      v16h bfrag = frag_from(bp, bp + 16);
      acc[f] = WMMA_F32_F16(afrag, bfrag, acc[f]);
    }
  }

  const int c  = lane & 15;
  const int hi = (lane & 16) ? 8 : 0;

#pragma unroll
  for (int f = 0; f < 4; ++f) {
    const float bv_ = bias[h * 64 + f * 16 + c];
#pragma unroll
    for (int r = 0; r < 8; ++r) acc[f][r] += bv_;
  }

  if (mat < 2) {
    _Float16* out = (mat == 0) ? qf : kf;
#pragma unroll
    for (int f = 0; f < 2; ++f) {  // RoPE pairs: (d, d+32), freq j = d
      const int j = f * 16 + c;
      const float invf = __powf(10000.0f, -(float)j * (1.0f / 32.0f));
#pragma unroll
      for (int r = 0; r < 8; ++r) {
        const int s = (rows + r + hi) & 2047;
        const float ang = (float)s * invf;
        const float cs = __cosf(ang), sn = __sinf(ang);
        const float x1 = acc[f][r], x2 = acc[f + 2][r];
        acc[f][r]     = x1 * cs - x2 * sn;
        acc[f + 2][r] = x1 * sn + x2 * cs;
      }
    }
#pragma unroll
    for (int f = 0; f < 4; ++f) {
      const int d = f * 16 + c;
#pragma unroll
      for (int r = 0; r < 8; ++r) {
        const int tok = rows + r + hi;
        const int b = tok >> 11, s = tok & 2047;
        out[(((size_t)(b * 16 + h)) * 2048 + s) * 64 + d] = (_Float16)acc[f][r];
      }
    }
  } else {  // V transposed: [b,h,d,s]
#pragma unroll
    for (int f = 0; f < 4; ++f) {
      const int d = f * 16 + c;
#pragma unroll
      for (int r = 0; r < 8; ++r) {
        const int tok = rows + r + hi;
        const int b = tok >> 11, s = tok & 2047;
        vf[(((size_t)(b * 16 + h)) * 64 + d) * 2048 + s] = (_Float16)acc[f][r];
      }
    }
  }
}

// ---------------------------------------------------------------------------
// Kernel 2: causal flash attention (online softmax), one wave = 16 q rows.
//   grid = (S/64) * B*NH = 32*32 = 1024 blocks of 128 (4 independent waves)
//   Per 32-key chunk: 4 WMMAs for scores, LDS transpose of P, 4 WMMAs for O.
// ---------------------------------------------------------------------------
__global__ __launch_bounds__(128) void flash_attn_kernel(
    const _Float16* __restrict__ qf, const _Float16* __restrict__ kf,
    const _Float16* __restrict__ vf, _Float16* __restrict__ attn) {
  __shared__ _Float16 pbuf[4][16 * 32];

  const int qb   = blockIdx.x & 31;
  const int bh   = blockIdx.x >> 5;
  const int b    = bh >> 4, h = bh & 15;
  const int wave = threadIdx.x >> 5;
  const int lane = threadIdx.x & 31;
  const int q0   = qb * 64 + wave * 16;

  const _Float16* qbase = qf + (size_t)bh * 2048 * 64;
  const _Float16* kbase = kf + (size_t)bh * 2048 * 64;
  const _Float16* vbase = vf + (size_t)bh * 64 * 2048;
  _Float16* pb = pbuf[wave];

  const int c  = lane & 15;
  const int hi = (lane & 16) ? 8 : 0;

  // Q A-fragments, d-chunks {0,32}
  v16h qa[2];
  {
    const int qrow = q0 + c;
#pragma unroll
    for (int cc = 0; cc < 2; ++cc) {
      const _Float16* p = qbase + (size_t)qrow * 64 + cc * 32 + ((lane & 16) ? 8 : 0);
      qa[cc] = frag_from(p, p + 16);
    }
  }

  v8f o[4] = {};
  float mrun[8], lrun[8];
#pragma unroll
  for (int r = 0; r < 8; ++r) { mrun[r] = -1e30f; lrun[r] = 0.0f; }

  for (int kt = 0; kt <= q0 + 15; kt += 32) {
    // ---- scores S = Q Kt / sqrt(d), two 16-key groups ----
    v8f s0 = {}, s1 = {};
#pragma unroll
    for (int cc = 0; cc < 2; ++cc) {
      {
        const _Float16* p = kbase + (size_t)(kt + c) * 64 + cc * 32 + ((lane & 16) ? 8 : 0);
        v16h kb = frag_from(p, p + 16);
        s0 = WMMA_F32_F16(qa[cc], kb, s0);
      }
      {
        const _Float16* p = kbase + (size_t)(kt + 16 + c) * 64 + cc * 32 + ((lane & 16) ? 8 : 0);
        v16h kb = frag_from(p, p + 16);
        s1 = WMMA_F32_F16(qa[cc], kb, s1);
      }
    }
#pragma unroll
    for (int r = 0; r < 8; ++r) { s0[r] *= 0.125f; s1[r] *= 0.125f; }

    if (kt + 31 > q0) {  // analytic causal mask, only near the diagonal
#pragma unroll
      for (int r = 0; r < 8; ++r) {
        const int qg = q0 + r + hi;
        if (kt + c      > qg) s0[r] = -1e9f;
        if (kt + 16 + c > qg) s1[r] = -1e9f;
      }
    }

    // ---- online softmax (rows 0-7 in lanes 0-15, rows 8-15 in lanes 16-31) ----
    float sc[8];
#pragma unroll
    for (int r = 0; r < 8; ++r) {
      float lm = fmaxf(s0[r], s1[r]);
#pragma unroll
      for (int m = 8; m >= 1; m >>= 1) lm = fmaxf(lm, __shfl_xor(lm, m, 32));
      const float mn = fmaxf(mrun[r], lm);
      sc[r]   = __expf(mrun[r] - mn);
      mrun[r] = mn;
      const float p0 = __expf(s0[r] - mn);
      const float p1 = __expf(s1[r] - mn);
      pb[(r + hi) * 32 + c]      = (_Float16)p0;
      pb[(r + hi) * 32 + 16 + c] = (_Float16)p1;
      float ps = p0 + p1;
#pragma unroll
      for (int m = 8; m >= 1; m >>= 1) ps += __shfl_xor(ps, m, 32);
      lrun[r] = lrun[r] * sc[r] + ps;
    }
#pragma unroll
    for (int dc = 0; dc < 4; ++dc)
#pragma unroll
      for (int r = 0; r < 8; ++r) o[dc][r] *= sc[r];

    // ---- P as A-fragment via per-wave LDS transpose ----
    const _Float16* pp = pb + (size_t)c * 32 + ((lane & 16) ? 8 : 0);
    v16h pfrag = frag_from(pp, pp + 16);

    // ---- O += P V, V is [d][s] so keys are contiguous per lane ----
#pragma unroll
    for (int dc = 0; dc < 4; ++dc) {
      const _Float16* vp = vbase + (size_t)(dc * 16 + c) * 2048 + kt + ((lane & 16) ? 8 : 0);
      v16h vfr = frag_from(vp, vp + 16);
      o[dc] = WMMA_F32_F16(pfrag, vfr, o[dc]);
    }
  }

  // normalize and store attn as fp16 [b, s, h*64+d] for the output GEMM
#pragma unroll
  for (int r = 0; r < 8; ++r) {
    const float inv = 1.0f / lrun[r];
    const int qg = q0 + r + hi;
#pragma unroll
    for (int dc = 0; dc < 4; ++dc)
      attn[((size_t)(b * 2048 + qg)) * 1024 + h * 64 + dc * 16 + c] = (_Float16)(o[dc][r] * inv);
  }
}

// ---------------------------------------------------------------------------
// Kernel 3: output projection attn(4096x1024) @ wo(1024x1024) + bo -> fp32
//   grid = 32 row-blocks * 16 col-blocks = 512 blocks of 256 (8 waves)
// ---------------------------------------------------------------------------
__global__ __launch_bounds__(256) void out_proj_kernel(
    const _Float16* __restrict__ attn, const float* __restrict__ wo,
    const float* __restrict__ bo, float* __restrict__ out) {
  __shared__ _Float16 wt[64 * 40];

  const int nb   = blockIdx.x & 15;
  const int rbm  = blockIdx.x >> 4;
  const int wave = threadIdx.x >> 5;
  const int lane = threadIdx.x & 31;
  const int rows = rbm * 128 + wave * 16;
  const int n0   = nb * 64;

  v8f acc[4] = {};
  const int arow = rows + (lane & 15);
  const int aoff = (lane & 16) ? 8 : 0;

  for (int k0 = 0; k0 < 1024; k0 += 32) {
    __syncthreads();
    {
      const int nl = threadIdx.x & 63;
      const int kg = threadIdx.x >> 6;
#pragma unroll
      for (int i = 0; i < 8; ++i) {
        const int kl = kg * 8 + i;
        wt[nl * 40 + kl] = (_Float16)wo[(size_t)(k0 + kl) * 1024 + n0 + nl];
      }
      if (k0 + 32 < 1024)
        __builtin_prefetch(&wo[(size_t)(k0 + 32) * 1024 + n0 + nl], 0, 1);
    }
    __syncthreads();

    const _Float16* ap = attn + (size_t)arow * 1024 + k0 + aoff;
    v16h afrag = frag_from(ap, ap + 16);

#pragma unroll
    for (int f = 0; f < 4; ++f) {
      const _Float16* bp = &wt[(f * 16 + (lane & 15)) * 40 + ((lane & 16) ? 8 : 0)];
      v16h bfrag = frag_from(bp, bp + 16);
      acc[f] = WMMA_F32_F16(afrag, bfrag, acc[f]);
    }
  }

  const int c  = lane & 15;
  const int hi = (lane & 16) ? 8 : 0;
#pragma unroll
  for (int f = 0; f < 4; ++f) {
    const float bv_ = bo[n0 + f * 16 + c];
#pragma unroll
    for (int r = 0; r < 8; ++r)
      out[(size_t)(rows + r + hi) * 1024 + n0 + f * 16 + c] = acc[f][r] + bv_;
  }
}

// ---------------------------------------------------------------------------
extern "C" void kernel_launch(void* const* d_in, const int* in_sizes, int n_in,
                              void* d_out, int out_size, void* d_ws, size_t ws_size,
                              hipStream_t stream) {
  const float* hidden = (const float*)d_in[0];
  // d_in[1] = attention_mask: causal, computed analytically (never read)
  const float* wq = (const float*)d_in[2];
  const float* bq = (const float*)d_in[3];
  const float* wk = (const float*)d_in[4];
  const float* bk = (const float*)d_in[5];
  const float* wv = (const float*)d_in[6];
  const float* bv = (const float*)d_in[7];
  const float* wo = (const float*)d_in[8];
  const float* bo = (const float*)d_in[9];
  float* out = (float*)d_out;

  const size_t N = (size_t)2 * 16 * 2048 * 64;  // B*NH*S*HD elements
  _Float16* qf = (_Float16*)d_ws;
  _Float16* kf = qf + N;
  _Float16* vf = kf + N;
  _Float16* at = vf + N;

  qkv_rope_kernel<<<1536, 256, 0, stream>>>(hidden, wq, wk, wv, bq, bk, bv, qf, kf, vf);
  flash_attn_kernel<<<1024, 128, 0, stream>>>(qf, kf, vf, at);
  out_proj_kernel<<<512, 256, 0, stream>>>(at, wo, bo, out);
}
